// OldLRU_81595788689679
// MI455X (gfx1250) — matrix operand order repaired and checked by reading, not compiled, not run on hardware
//
#include <hip/hip_runtime.h>
#include <hip/hip_bf16.h>

typedef __attribute__((ext_vector_type(16))) __bf16 v16bf;
typedef __attribute__((ext_vector_type(8)))  float  v8f;
typedef __attribute__((ext_vector_type(4)))  float  f32x4;
typedef __attribute__((ext_vector_type(8)))  __bf16 bf16x8;

#define BATCHN 8
#define LSEQ   8192
#define DOUTN  512
#define DHIDN  256
#define MROWS  (BATCHN * LSEQ)          // 65536
#define BM     128                      // rows per block
#define CHUNK  128                      // scan chunk length
#define NCHUNK (LSEQ / CHUNK)           // 64

// ---------------- workspace layout (bytes) ----------------
// All GEMM B-operands stored N-major (transposed): Bt[n][k], k contiguous.
static constexpr size_t OFF_BN   = 0;                                        // bf16 [512x512] B_norm^T, N-major
static constexpr size_t OFF_CF   = OFF_BN  + (size_t)DOUTN * DOUTN * 2;      // bf16 [512x512] [C_re^T; -C_im^T], N-major
static constexpr size_t OFF_LAM  = OFF_CF  + (size_t)DOUTN * DOUTN * 2;      // f32 [512] lam re|im
static constexpr size_t OFF_LAMP = OFF_LAM + 512 * 4;                        // f32 [512] lam^CHUNK re|im
static constexpr size_t OFF_E    = OFF_LAMP + 512 * 4;                       // f32 [8*64*512] chunk partials
static constexpr size_t OFF_CAR  = OFF_E   + (size_t)BATCHN * NCHUNK * 512 * 4; // f32 chunk carries
static constexpr size_t OFF_XPF  = OFF_CAR + (size_t)BATCHN * NCHUNK * 512 * 4; // f32 [65536x512] xp
static constexpr size_t OFF_XPB  = OFF_XPF + (size_t)MROWS * DOUTN * 4;      // bf16 [65536x512] xp
static constexpr size_t OFF_BU   = OFF_XPB + (size_t)MROWS * DOUTN * 2;      // bf16 [65536x512] Bu (re|im cols)
static constexpr size_t OFF_H    = OFF_BU  + (size_t)MROWS * DOUTN * 2;      // bf16 [65536x512] h  (re|im cols)

// ---------------- CDNA5 async global->LDS copy (ASYNCcnt-tracked) ----------------
// Moves 16 bytes per lane directly into LDS, bypassing VGPRs.
__device__ __forceinline__ void async_copy16(void* lds_dst, const void* gsrc) {
  unsigned off = (unsigned)(uintptr_t)lds_dst;   // LDS aperture: addr[31:0] = LDS byte offset
  asm volatile("global_load_async_to_lds_b128 %0, %1, off"
               :: "v"(off), "v"(gsrc)
               : "memory");
}
template <int N>
__device__ __forceinline__ void wait_async() {
  if constexpr (N == 0)      asm volatile("s_wait_asynccnt 0x0" ::: "memory");
  else if constexpr (N == 2) asm volatile("s_wait_asynccnt 0x2" ::: "memory");
  else                       asm volatile("s_wait_asynccnt 0x4" ::: "memory");
}

// ---------------- WMMA fragment loaders (ISA 7.12.2 layouts) ----------------
// A: 16x32 bf16 tile, row-major in LDS with given stride.
__device__ __forceinline__ v16bf load_a_frag(const __bf16* base, int stride) {
  int lane = threadIdx.x & 31;
  int m    = lane & 15;
  int koff = (lane >> 4) * 8;
  const __bf16* row = base + m * stride + koff;
  bf16x8 lo = *(const bf16x8*)(row);        // K = koff .. koff+7
  bf16x8 hi = *(const bf16x8*)(row + 16);   // K = koff+16 .. koff+23
  v16bf a;
#pragma unroll
  for (int q = 0; q < 8; ++q) { a[q] = lo[q]; a[8 + q] = hi[q]; }
  return a;
}

// B: 32x16 (K x N) tile stored N-major in LDS: tile base + n*32 + k, k contiguous.
__device__ __forceinline__ v16bf load_b_frag_t(const __bf16* base) {
  int lane  = threadIdx.x & 31;
  int n     = lane & 15;
  int kbase = (lane >> 4) * 8;
  const __bf16* p = base + n * 32 + kbase;
  bf16x8 lo = *(const bf16x8*)(p);
  bf16x8 hi = *(const bf16x8*)(p + 16);
  v16bf b;
#pragma unroll
  for (int q = 0; q < 8; ++q) { b[q] = lo[q]; b[8 + q] = hi[q]; }
  return b;
}

// ---------------- prep: lam, lam^128, packed BnT / CfT (N-major) ----------------
__global__ __launch_bounds__(256) void prep_kernel(
    const float* __restrict__ nu_log, const float* __restrict__ theta_log,
    const float* __restrict__ gamma_log,
    const float* __restrict__ B_re, const float* __restrict__ B_im,
    const float* __restrict__ C_re, const float* __restrict__ C_im,
    float* __restrict__ lam, float* __restrict__ lamp,
    __bf16* __restrict__ Bn, __bf16* __restrict__ Cf) {
  if (blockIdx.x == 0 && threadIdx.x < DHIDN) {
    int n = threadIdx.x;
    float mag = __expf(-__expf(nu_log[n]));
    float th  = __expf(theta_log[n]);
    float lre = mag * __cosf(th);
    float lim = mag * __sinf(th);
    lam[n] = lre; lam[256 + n] = lim;
    float pre = lre, pim = lim;
#pragma unroll
    for (int i = 0; i < 7; ++i) {           // lam^128 by repeated squaring
      float nre = pre * pre - pim * pim;
      float nim = 2.f * pre * pim;
      pre = nre; pim = nim;
    }
    lamp[n] = pre; lamp[256 + n] = pim;
  }
  int tid = blockIdx.x * blockDim.x + threadIdx.x;
  int nthreads = blockDim.x * gridDim.x;
  for (int i = tid; i < DOUTN * DOUTN; i += nthreads) {
    int j = i >> 9, k = i & 511;            // output col j, reduction dim k
    float bv;
    if (j < DHIDN) bv = B_re[j * DOUTN + k] * __expf(gamma_log[j]);
    else { int n2 = j - DHIDN; bv = B_im[n2 * DOUTN + k] * __expf(gamma_log[n2]); }
    Bn[(size_t)j * DOUTN + k] = (__bf16)bv;
    float cv;
    if (k < DHIDN) cv = C_re[j * DHIDN + k];
    else           cv = -C_im[j * DHIDN + (k - DHIDN)];
    Cf[(size_t)j * DOUTN + k] = (__bf16)cv;
  }
}

// repack W (f32 [512x512], K-major) -> bf16 N-major transpose: Wb[j][k] = W[k][j]
__global__ __launch_bounds__(256) void repack_w_kernel(const float* __restrict__ W,
                                                       __bf16* __restrict__ Wb) {
  int tid = blockIdx.x * blockDim.x + threadIdx.x;
  for (int i = tid; i < DOUTN * DOUTN; i += blockDim.x * gridDim.x) {
    int j = i >> 9, k = i & 511;
    Wb[(size_t)j * DOUTN + k] = (__bf16)W[(size_t)k * DOUTN + j];
  }
}

// ---------------- generic tiled bf16 WMMA GEMM, M=65536, N=K=512 ----------------
// Async double-buffered LDS staging; B operand (Bt) is N-major: Bt[n*512 + k].
// MODE 0: out_f32 = acc + vec[col]; out_bf16 = bf16(out_f32)           (xp = x@W + b)
// MODE 1: out_bf16 = bf16(acc)                                         (Bu = xp@Bn)
// MODE 2: out_f32 = acc + vec[col] * xp_in[row,col]                    (out = H@Cf + D*xp)
template <int MODE, bool AF32>
__global__ __launch_bounds__(256) void gemm_kernel(
    const void* __restrict__ A, const __bf16* __restrict__ Bt,
    const float* __restrict__ vec, const float* __restrict__ xp_in,
    float* __restrict__ out_f32, __bf16* __restrict__ out_bf16) {
  __shared__ __bf16 lA[2][BM * 32];         // 2 x 8 KB, [128 rows x 32 K] row-major
  __shared__ __bf16 lB[2][128 * 32];        // 2 x 8 KB, [128 cols x 32 K] N-major
  int tid  = threadIdx.x;
  int lane = tid & 31;
  int wave = tid >> 5;
  int wm   = wave & 3;                      // 4 waves along M (32 rows each)
  int wn   = wave >> 2;                     // 2 waves along N (64 cols each)
  size_t row0 = (size_t)blockIdx.x * BM;

  // per-thread staging coordinates (contiguous 16-element runs, 16B aligned)
  int ar = tid >> 1, ac = (tid & 1) * 16;   // A panel: row, K-half
  int bn = tid >> 1, bk = (tid & 1) * 16;   // B panel: col, K-half
  constexpr int NI = AF32 ? 2 : 4;          // async ops per thread per panel

  for (int nb = 0; nb < 4; ++nb) {          // N chunks of 128
    int ncol0 = nb * 128;
    v8f acc[2][4] = {};

    auto stage = [&](int kp, int buf) {
      int k0 = kp * 32;
      if constexpr (AF32) {                 // f32 A: convert in registers, sync LDS store
        const float* src = (const float*)A + (row0 + ar) * DOUTN + (k0 + ac);
        f32x4 s[4];
#pragma unroll
        for (int q = 0; q < 4; ++q) s[q] = *(const f32x4*)(src + 4 * q);
        bf16x8 d0, d1;
#pragma unroll
        for (int q = 0; q < 8; ++q) d0[q] = (__bf16)s[q >> 2][q & 3];
#pragma unroll
        for (int q = 0; q < 8; ++q) d1[q] = (__bf16)s[2 + (q >> 2)][q & 3];
        *(bf16x8*)&lA[buf][ar * 32 + ac]     = d0;
        *(bf16x8*)&lA[buf][ar * 32 + ac + 8] = d1;
      } else {                              // bf16 A: async global->LDS
        const __bf16* src = (const __bf16*)A + (row0 + ar) * DOUTN + (k0 + ac);
        async_copy16(&lA[buf][ar * 32 + ac],     src);
        async_copy16(&lA[buf][ar * 32 + ac + 8], src + 8);
      }
      const __bf16* bsrc = Bt + (size_t)(ncol0 + bn) * DOUTN + (k0 + bk);
      async_copy16(&lB[buf][bn * 32 + bk],     bsrc);
      async_copy16(&lB[buf][bn * 32 + bk + 8], bsrc + 8);
    };

    stage(0, 0);                            // prologue: panel 0 -> buffer 0
    for (int kp = 0; kp < 16; ++kp) {       // K panels of 32
      int cur = kp & 1;
      if (kp + 1 < 16) {
        stage(kp + 1, cur ^ 1);             // overlap: issue next panel
        wait_async<NI>();                   // current panel landed (in-order per wave)
      } else {
        wait_async<0>();
      }
      __syncthreads();
      v16bf afr[2], bfr[4];
      afr[0] = load_a_frag(&lA[cur][(wm * 32) * 32], 32);
      afr[1] = load_a_frag(&lA[cur][(wm * 32 + 16) * 32], 32);
#pragma unroll
      for (int j = 0; j < 4; ++j) bfr[j] = load_b_frag_t(&lB[cur][(wn * 64 + j * 16) * 32]);
#pragma unroll
      for (int i = 0; i < 2; ++i)
#pragma unroll
        for (int j = 0; j < 4; ++j)
          acc[i][j] = __builtin_amdgcn_wmma_f32_16x16x32_bf16(
              false, afr[i], false, bfr[j], (short)0, acc[i][j], false, false);
      __syncthreads();                      // protect buffer reuse two iterations later
    }
    // epilogue: C/D layout -> row = 8*(lane>=16)+v, col = lane&15
    int cl = lane & 15, hi = lane >> 4;
#pragma unroll
    for (int i = 0; i < 2; ++i)
#pragma unroll
      for (int j = 0; j < 4; ++j)
#pragma unroll
        for (int v = 0; v < 8; ++v) {
          int r = wm * 32 + i * 16 + hi * 8 + v;
          int c = ncol0 + wn * 64 + j * 16 + cl;
          size_t gi = (row0 + r) * DOUTN + c;
          float val = acc[i][j][v];
          if constexpr (MODE == 0) {
            val += vec[c];
            out_f32[gi]  = val;
            out_bf16[gi] = (__bf16)val;
          } else if constexpr (MODE == 1) {
            out_bf16[gi] = (__bf16)val;
          } else {
            out_f32[gi] = val + vec[c] * xp_in[gi];
          }
        }
  }
}

// ---------------- scan phase 1: per-chunk partial (h with zero seed) ----------------
__global__ __launch_bounds__(256) void scan_phase1(
    const __bf16* __restrict__ Bu, const float* __restrict__ lam, float* __restrict__ e) {
  int bc = blockIdx.x;                      // 0..511 = b*64 + chunk
  int b = bc >> 6, c = bc & 63;
  int n = threadIdx.x;                      // channel 0..255
  float lre = lam[n], lim = lam[256 + n];
  float hre = 0.f, him = 0.f;
  const __bf16* p = Bu + ((size_t)b * LSEQ + (size_t)c * CHUNK) * DOUTN;
  for (int t = 0; t < CHUNK; ++t) {
    float bre = (float)p[n];
    float bim = (float)p[256 + n];
    float nre = lre * hre - lim * him + bre;
    float nim = lre * him + lim * hre + bim;
    hre = nre; him = nim;
    p += DOUTN;
  }
  e[(size_t)bc * 512 + n]       = hre;
  e[(size_t)bc * 512 + 256 + n] = him;
}

// ---------------- scan phase 2: carry chain over chunks with lam^CHUNK ----------------
__global__ __launch_bounds__(256) void scan_phase2(
    const float* __restrict__ e, const float* __restrict__ lamp, float* __restrict__ carry) {
  int b = blockIdx.x;                       // 0..7
  int n = threadIdx.x;                      // 0..255
  float pre = lamp[n], pim = lamp[256 + n];
  float cre = 0.f, cim = 0.f;
  for (int c = 0; c < NCHUNK; ++c) {
    size_t idx = (size_t)(b * NCHUNK + c) * 512;
    carry[idx + n]       = cre;             // h entering chunk c
    carry[idx + 256 + n] = cim;
    float ere = e[idx + n], eim = e[idx + 256 + n];
    float nre = pre * cre - pim * cim + ere;
    float nim = pre * cim + pim * cre + eim;
    cre = nre; cim = nim;
  }
}

// ---------------- scan phase 3: seeded re-scan, emit H (bf16) + final state ----------------
__global__ __launch_bounds__(256) void scan_phase3(
    const __bf16* __restrict__ Bu, const float* __restrict__ lam,
    const float* __restrict__ carry, __bf16* __restrict__ H, float* __restrict__ hlast) {
  int bc = blockIdx.x;
  int b = bc >> 6, c = bc & 63;
  int n = threadIdx.x;
  float lre = lam[n], lim = lam[256 + n];
  size_t cidx = (size_t)bc * 512;
  float hre = carry[cidx + n], him = carry[cidx + 256 + n];
  size_t rowbase = ((size_t)b * LSEQ + (size_t)c * CHUNK) * DOUTN;
  const __bf16* p = Bu + rowbase;
  __bf16* q = H + rowbase;
  for (int t = 0; t < CHUNK; ++t) {
    float bre = (float)p[n];
    float bim = (float)p[256 + n];
    float nre = lre * hre - lim * him + bre;
    float nim = lre * him + lim * hre + bim;
    hre = nre; him = nim;
    q[n]       = (__bf16)hre;
    q[256 + n] = (__bf16)him;
    p += DOUTN; q += DOUTN;
  }
  if (c == NCHUNK - 1) {                    // h[:, -1] as complex64 (re,im interleaved)
    hlast[(b * DHIDN + n) * 2 + 0] = hre;
    hlast[(b * DHIDN + n) * 2 + 1] = him;
  }
}

// ---------------- host entry ----------------
extern "C" void kernel_launch(void* const* d_in, const int* in_sizes, int n_in,
                              void* d_out, int out_size, void* d_ws, size_t ws_size,
                              hipStream_t stream) {
  const float* x         = (const float*)d_in[0];
  const float* W         = (const float*)d_in[1];
  const float* bvec      = (const float*)d_in[2];
  const float* nu_log    = (const float*)d_in[3];
  const float* theta_log = (const float*)d_in[4];
  const float* gamma_log = (const float*)d_in[5];
  const float* B_re      = (const float*)d_in[6];
  const float* B_im      = (const float*)d_in[7];
  const float* C_re      = (const float*)d_in[8];
  const float* C_im      = (const float*)d_in[9];
  const float* Dvec      = (const float*)d_in[10];

  char* ws = (char*)d_ws;
  __bf16* Bn    = (__bf16*)(ws + OFF_BN);
  __bf16* Cf    = (__bf16*)(ws + OFF_CF);
  float*  lam   = (float*)(ws + OFF_LAM);
  float*  lamp  = (float*)(ws + OFF_LAMP);
  float*  e     = (float*)(ws + OFF_E);
  float*  carry = (float*)(ws + OFF_CAR);
  float*  xpf   = (float*)(ws + OFF_XPF);
  __bf16* xpb   = (__bf16*)(ws + OFF_XPB);
  __bf16* Bu    = (__bf16*)(ws + OFF_BU);
  __bf16* H     = (__bf16*)(ws + OFF_H);

  float* out   = (float*)d_out;
  float* hlast = out + (size_t)MROWS * DOUTN;

  // weights/constants prep (all B-operands stored N-major)
  prep_kernel<<<64, 256, 0, stream>>>(nu_log, theta_log, gamma_log,
                                      B_re, B_im, C_re, C_im, lam, lamp, Bn, Cf);
  // W -> bf16 transposed, staged into the H region (consumed before phase 3 writes H).
  repack_w_kernel<<<256, 256, 0, stream>>>(W, H);
  // xp = x@W + b          (fp32 + bf16 copies)
  gemm_kernel<0, true><<<MROWS / BM, 256, 0, stream>>>(x, H, bvec, nullptr, xpf, xpb);
  // Bu = xp@Bn            (complex packed as re|im columns, bf16)
  gemm_kernel<1, false><<<MROWS / BM, 256, 0, stream>>>(xpb, Bn, nullptr, nullptr, nullptr, Bu);
  // chunked linear recurrence (lambda is time-constant)
  scan_phase1<<<BATCHN * NCHUNK, 256, 0, stream>>>(Bu, lam, e);
  scan_phase2<<<BATCHN, 256, 0, stream>>>(e, lamp, carry);
  scan_phase3<<<BATCHN * NCHUNK, 256, 0, stream>>>(Bu, lam, carry, H, hlast);
  // out = Re(h@C^T) + D*xp
  gemm_kernel<2, false><<<MROWS / BM, 256, 0, stream>>>(H, Cf, Dvec, xpf, out, nullptr);

  (void)in_sizes; (void)n_in; (void)out_size; (void)ws_size;
}
